// MCxC_GNN_Sensor_39281770890011
// MI455X (gfx1250) — compile-verified, matrix operand order checked by reading, not run on hardware
//
#include <hip/hip_runtime.h>
#include <hip/hip_bf16.h>

typedef __attribute__((ext_vector_type(16))) _Float16 v16h;
typedef __attribute__((ext_vector_type(8)))  float    v8f;

#define N_NODES 50000
#define N_EDGES 800000
#define NB      64
#define MM      50
#define FNODE   12
#define TE_CH   64
#define HDIM    128
#define K1REAL  76
#define K1PAD   96

// ---------------------------------------------------------------- utilities
__global__ void zero_kernel(float* __restrict__ p, int n) {
    int i = blockIdx.x * blockDim.x + threadIdx.x;
    if (i < n) p[i] = 0.0f;
}

// ---------------------------------------------------------------- conv1 + relu
// topo_map (B,1,50,50) -> t1 (B,16,50,50)
__global__ void conv1_kernel(const float* __restrict__ tmap,
                             const float* __restrict__ ck1,
                             const float* __restrict__ cb1,
                             float* __restrict__ t1) {
    int tid = blockIdx.x * blockDim.x + threadIdx.x;
    const int total = NB * 16 * MM * MM;
    if (tid >= total) return;
    int x  = tid % MM;
    int y  = (tid / MM) % MM;
    int oc = (tid / (MM * MM)) % 16;
    int b  = tid / (16 * MM * MM);
    float s = cb1[oc];
    const float* in = tmap + (size_t)b * MM * MM;
    #pragma unroll
    for (int ky = 0; ky < 3; ky++) {
        int yy = y + ky - 1;
        if ((unsigned)yy >= MM) continue;
        #pragma unroll
        for (int kx = 0; kx < 3; kx++) {
            int xx = x + kx - 1;
            if ((unsigned)xx >= MM) continue;
            s += in[yy * MM + xx] * ck1[oc * 9 + ky * 3 + kx];
        }
    }
    t1[tid] = fmaxf(s, 0.0f);
}

// ---------------------------------------------------------------- conv2 + spatial mean
// t1 (B,16,50,50) -> tfeat (B,64) ; block per (b,oc), 256 threads
__global__ void conv2_mean_kernel(const float* __restrict__ t1,
                                  const float* __restrict__ ck2,
                                  const float* __restrict__ cb2,
                                  float* __restrict__ tfeat) {
    __shared__ float wsh[144];
    __shared__ float red[256];
    int b  = blockIdx.x >> 6;
    int oc = blockIdx.x & 63;
    if (threadIdx.x < 144) wsh[threadIdx.x] = ck2[oc * 144 + threadIdx.x];
    __syncthreads();
    float s = 0.0f;
    for (int p = threadIdx.x; p < MM * MM; p += blockDim.x) {
        int y = p / MM, x = p % MM;
        float acc = 0.0f;
        for (int ic = 0; ic < 16; ic++) {
            const float* in = t1 + ((size_t)(b * 16 + ic)) * MM * MM;
            const float* wk = wsh + ic * 9;
            #pragma unroll
            for (int ky = 0; ky < 3; ky++) {
                int yy = y + ky - 1;
                if ((unsigned)yy >= MM) continue;
                #pragma unroll
                for (int kx = 0; kx < 3; kx++) {
                    int xx = x + kx - 1;
                    if ((unsigned)xx >= MM) continue;
                    acc += in[yy * MM + xx] * wk[ky * 3 + kx];
                }
            }
        }
        s += acc;
    }
    red[threadIdx.x] = s;
    __syncthreads();
    for (int off = 128; off > 0; off >>= 1) {
        if ((int)threadIdx.x < off) red[threadIdx.x] += red[threadIdx.x + off];
        __syncthreads();
    }
    if (threadIdx.x == 0)
        tfeat[blockIdx.x] = red[0] * (1.0f / (MM * MM)) + cb2[oc];
}

// ---------------------------------------------------------------- build concat feature in f16 (N x 96)
__global__ void build_xh_kernel(const float* __restrict__ xn,
                                const int* __restrict__ batch,
                                const float* __restrict__ tfeat,
                                _Float16* __restrict__ xh) {
    int tid = blockIdx.x * blockDim.x + threadIdx.x;
    const int total = N_NODES * K1PAD;
    if (tid >= total) return;
    int k = tid % K1PAD;
    int n = tid / K1PAD;
    float v;
    if (k < FNODE)       v = xn[n * FNODE + k];
    else if (k < K1REAL) v = tfeat[batch[n] * TE_CH + (k - FNODE)];
    else                 v = 0.0f;
    xh[tid] = (_Float16)v;
}

// ---------------------------------------------------------------- pack W (K x 128, f32) into WMMA B-fragment layout (f16)
// Wf[((kc*8 + t)*32 + lane)*16 + j] = W[kc*32 + 16*(lane>>4) + j][t*16 + (lane&15)]
__global__ void pack_w_kernel(const float* __restrict__ W,
                              _Float16* __restrict__ Wf, int K, int KC) {
    int tid = blockIdx.x * blockDim.x + threadIdx.x;
    int total = KC * 8 * 32 * 16;
    if (tid >= total) return;
    int j    = tid & 15;
    int lane = (tid >> 4) & 31;
    int t    = (tid >> 9) & 7;
    int c    = tid >> 12;
    int r = lane & 15, g = lane >> 4;
    int k   = c * 32 + g * 16 + j;
    int col = t * 16 + r;
    Wf[tid] = (k < K) ? (_Float16)W[k * HDIM + col] : (_Float16)0.0f;
}

// ---------------------------------------------------------------- degree / norm
__global__ void deg_kernel(const int* __restrict__ dst, float* __restrict__ deg) {
    int e = blockIdx.x * blockDim.x + threadIdx.x;
    if (e < N_EDGES) atomicAdd(&deg[dst[e]], 1.0f);
}
__global__ void norm_kernel(float* __restrict__ degnorm) {
    int i = blockIdx.x * blockDim.x + threadIdx.x;
    if (i < N_NODES) degnorm[i] = rsqrtf(degnorm[i] + 1.0f);
}

// ---------------------------------------------------------------- WMMA GEMM: H (N x 128) = A (N x Ks, f16) @ Wpacked
// Packed weights (KC*8KB <= 32KB) staged in LDS once per block, shared by all
// 8 waves; WMMAs consume B fragments via ds_load_b128. A fragments for all
// k-chunks hoisted into registers so global loads issue as one clause.
// One wave -> 16 rows x 128 cols; uniform wave guard keeps EXEC all-1s.
template <int KC>
__global__ void gemm_wmma_kernel(const _Float16* __restrict__ A,
                                 const _Float16* __restrict__ Wf,
                                 float* __restrict__ Hout,
                                 int Ks) {
    __shared__ _Float16 wlds[KC * 4096];

    // cooperative 128-bit copy of packed weights into LDS
    {
        const uint4* gsrc = (const uint4*)Wf;
        uint4* ldst = (uint4*)wlds;
        const int nvec = KC * 4096 / 8;   // 8 halves per uint4
        for (int i = threadIdx.x; i < nvec; i += blockDim.x) ldst[i] = gsrc[i];
    }
    __syncthreads();

    int wave = threadIdx.x >> 5;
    int lane = threadIdx.x & 31;
    int rowTile = blockIdx.x * (blockDim.x >> 5) + wave;
    if (rowTile >= N_NODES / 16) return;
    int r = lane & 15, g = lane >> 4;

    // hoist all A fragments (global loads cluster together)
    const _Float16* arow = A + (size_t)(rowTile * 16 + r) * Ks + g * 16;
    v16h afrag[KC];
    #pragma unroll
    for (int kc = 0; kc < KC; kc++) afrag[kc] = *(const v16h*)(arow + kc * 32);

    v8f acc[8];
    v8f zero = {0.f, 0.f, 0.f, 0.f, 0.f, 0.f, 0.f, 0.f};
    #pragma unroll
    for (int t = 0; t < 8; t++) acc[t] = zero;

    #pragma unroll
    for (int kc = 0; kc < KC; kc++) {
        #pragma unroll
        for (int t = 0; t < 8; t++) {
            v16h b = *(const v16h*)(wlds + (size_t)(kc * 8 + t) * 512 + lane * 16);
            acc[t] = __builtin_amdgcn_wmma_f32_16x16x32_f16(
                false, afrag[kc], false, b, (short)0, acc[t], false, false);
        }
    }

    int rowBase = rowTile * 16 + g * 8;
    #pragma unroll
    for (int t = 0; t < 8; t++) {
        #pragma unroll
        for (int j = 0; j < 8; j++) {
            Hout[(size_t)(rowBase + j) * HDIM + t * 16 + r] = acc[t][j];
        }
    }
}

// ---------------------------------------------------------------- edge aggregation: agg[dst] += h[src] * norm[src]*norm[dst]
// 32 threads per edge, float4 per thread; h/agg resident in 192MB L2
__global__ void edge_agg_kernel(const float* __restrict__ h,
                                const float* __restrict__ nrm,
                                const int* __restrict__ src,
                                const int* __restrict__ dst,
                                float* __restrict__ agg) {
    int idx = blockIdx.x * blockDim.x + threadIdx.x;
    if (idx >= N_EDGES * 32) return;
    int e = idx >> 5;
    int q = (idx & 31) << 2;
    int s = src[e], d = dst[e];
    float c = nrm[s] * nrm[d];
    const float4 hv = *(const float4*)(h + (size_t)s * HDIM + q);
    float* ap = agg + (size_t)d * HDIM + q;
    atomicAdd(ap + 0, hv.x * c);
    atomicAdd(ap + 1, hv.y * c);
    atomicAdd(ap + 2, hv.z * c);
    atomicAdd(ap + 3, hv.w * c);
}

// ---------------------------------------------------------------- post layer: x = relu(agg + h*norm^2 + b)  -> f16
__global__ void post_layer_kernel(const float* __restrict__ agg,
                                  const float* __restrict__ h,
                                  const float* __restrict__ nrm,
                                  const float* __restrict__ bias,
                                  _Float16* __restrict__ xout) {
    int tid = blockIdx.x * blockDim.x + threadIdx.x;
    const int total = N_NODES * HDIM;
    if (tid >= total) return;
    int f = tid & (HDIM - 1);
    int n = tid >> 7;
    float nn = nrm[n];
    float v = agg[tid] + h[tid] * nn * nn + bias[f];
    xout[tid] = (_Float16)fmaxf(v, 0.0f);
}

// ---------------------------------------------------------------- finalize: node_out = relu(gcn2) @ Wd + bd
__global__ void finalize_kernel(const float* __restrict__ agg,
                                const float* __restrict__ h,
                                const float* __restrict__ nrm,
                                const float* __restrict__ b2,
                                const float* __restrict__ Wd,
                                const float* __restrict__ bd,
                                float* __restrict__ node_out) {
    int n = blockIdx.x * blockDim.x + threadIdx.x;
    if (n >= N_NODES) return;
    float nn = nrm[n]; nn *= nn;
    float acc = 0.0f;
    for (int f = 0; f < HDIM; f++) {
        float v = agg[(size_t)n * HDIM + f] + h[(size_t)n * HDIM + f] * nn + b2[f];
        acc += fmaxf(v, 0.0f) * Wd[f];
    }
    node_out[n] = acc + bd[0];
}

// ---------------------------------------------------------------- scatter node_out into grid
__global__ void scatter_kernel(const float* __restrict__ xn,
                               const int* __restrict__ batch,
                               const float* __restrict__ node_out,
                               float* __restrict__ out) {
    int n = blockIdx.x * blockDim.x + threadIdx.x;
    if (n >= N_NODES) return;
    float cx = xn[n * FNODE + 1];
    float cy = xn[n * FNODE + 2];
    int xi = (int)(cx * (float)(MM - 1)); xi = min(max(xi, 0), MM - 1);
    int yi = (int)(cy * (float)(MM - 1)); yi = min(max(yi, 0), MM - 1);
    atomicAdd(&out[batch[n] * (MM * MM) + yi * MM + xi], node_out[n]);
}

// ---------------------------------------------------------------- apply mask
__global__ void mask_kernel(const float* __restrict__ mask, float* __restrict__ out) {
    int tid = blockIdx.x * blockDim.x + threadIdx.x;
    const int total = NB * MM * MM;
    if (tid >= total) return;
    out[tid] *= mask[tid % (MM * MM)];
}

// ================================================================ launch
static inline int cdiv(int a, int b) { return (a + b - 1) / b; }

extern "C" void kernel_launch(void* const* d_in, const int* in_sizes, int n_in,
                              void* d_out, int out_size, void* d_ws, size_t ws_size,
                              hipStream_t stream) {
    const float* x_nodes = (const float*)d_in[0];
    const int*   ei      = (const int*)  d_in[1];
    const int*   batch   = (const int*)  d_in[2];
    const float* tmap    = (const float*)d_in[3];
    const float* ck1     = (const float*)d_in[4];
    const float* cb1     = (const float*)d_in[5];
    const float* ck2     = (const float*)d_in[6];
    const float* cb2     = (const float*)d_in[7];
    const float* W1      = (const float*)d_in[8];
    const float* b1      = (const float*)d_in[9];
    const float* W2      = (const float*)d_in[10];
    const float* b2      = (const float*)d_in[11];
    const float* Wd      = (const float*)d_in[12];
    const float* bd      = (const float*)d_in[13];
    const float* mask    = (const float*)d_in[14];
    float* out = (float*)d_out;

    const int* src = ei;
    const int* dst = ei + N_EDGES;

    // -------- workspace carve (256B aligned)
    char* base = (char*)d_ws;
    size_t off = 0;
    auto carve = [&](size_t bytes) -> void* {
        off = (off + 255) & ~(size_t)255;
        void* p = base + off;
        off += bytes;
        return p;
    };
    float*    t1       = (float*)   carve((size_t)NB * 16 * MM * MM * 4);
    float*    tfeat    = (float*)   carve((size_t)NB * TE_CH * 4);
    _Float16* xh       = (_Float16*)carve((size_t)N_NODES * K1PAD * 2);
    _Float16* W1f      = (_Float16*)carve((size_t)3 * 4096 * 2);
    _Float16* W2f      = (_Float16*)carve((size_t)4 * 4096 * 2);
    float*    degnorm  = (float*)   carve((size_t)N_NODES * 4);
    float*    hbuf     = (float*)   carve((size_t)N_NODES * HDIM * 4);
    float*    aggbuf   = (float*)   carve((size_t)N_NODES * HDIM * 4);
    _Float16* x2h      = (_Float16*)carve((size_t)N_NODES * HDIM * 2);
    float*    node_out = (float*)   carve((size_t)N_NODES * 4);

    const int BT = 256;

    // -------- topo branch
    conv1_kernel<<<cdiv(NB * 16 * MM * MM, BT), BT, 0, stream>>>(tmap, ck1, cb1, t1);
    conv2_mean_kernel<<<NB * TE_CH, BT, 0, stream>>>(t1, ck2, cb2, tfeat);

    // -------- features + packed weights
    build_xh_kernel<<<cdiv(N_NODES * K1PAD, BT), BT, 0, stream>>>(x_nodes, batch, tfeat, xh);
    pack_w_kernel<<<cdiv(3 * 4096, BT), BT, 0, stream>>>(W1, W1f, K1REAL, 3);
    pack_w_kernel<<<cdiv(4 * 4096, BT), BT, 0, stream>>>(W2, W2f, HDIM, 4);

    // -------- degree norm
    zero_kernel<<<cdiv(N_NODES, BT), BT, 0, stream>>>(degnorm, N_NODES);
    deg_kernel<<<cdiv(N_EDGES, BT), BT, 0, stream>>>(dst, degnorm);
    norm_kernel<<<cdiv(N_NODES, BT), BT, 0, stream>>>(degnorm);

    // -------- GCN layer 1
    gemm_wmma_kernel<3><<<cdiv(N_NODES / 16, 8), BT, 0, stream>>>(xh, W1f, hbuf, K1PAD);
    zero_kernel<<<cdiv(N_NODES * HDIM, BT), BT, 0, stream>>>(aggbuf, N_NODES * HDIM);
    edge_agg_kernel<<<cdiv(N_EDGES * 32, BT), BT, 0, stream>>>(hbuf, degnorm, src, dst, aggbuf);
    post_layer_kernel<<<cdiv(N_NODES * HDIM, BT), BT, 0, stream>>>(aggbuf, hbuf, degnorm, b1, x2h);

    // -------- GCN layer 2
    gemm_wmma_kernel<4><<<cdiv(N_NODES / 16, 8), BT, 0, stream>>>(x2h, W2f, hbuf, HDIM);
    zero_kernel<<<cdiv(N_NODES * HDIM, BT), BT, 0, stream>>>(aggbuf, N_NODES * HDIM);
    edge_agg_kernel<<<cdiv(N_EDGES * 32, BT), BT, 0, stream>>>(hbuf, degnorm, src, dst, aggbuf);

    // -------- head + scatter
    finalize_kernel<<<cdiv(N_NODES, BT), BT, 0, stream>>>(aggbuf, hbuf, degnorm, b2, Wd, bd, node_out);
    zero_kernel<<<cdiv(NB * MM * MM, BT), BT, 0, stream>>>(out, NB * MM * MM);
    scatter_kernel<<<cdiv(N_NODES, BT), BT, 0, stream>>>(x_nodes, batch, node_out, out);
    mask_kernel<<<cdiv(NB * MM * MM, BT), BT, 0, stream>>>(mask, out);
}
